// MFBasedModel_84653805404333
// MI455X (gfx1250) — matrix operand order repaired and checked by reading, not compiled, not run on hardware
//
#include <hip/hip_runtime.h>
#include <hip/hip_bf16.h>

#define EMB     32
#define BATCH   128
#define UID     200000
#define TOPK    50000
#define NTILES  (UID / 16)      // 12500
#define MTILES  (BATCH / 16)    // 8
#define CAP     1024
#define GEMM_BLOCKS ((NTILES + 7) / 8)   // 1563, 8 waves per 256-thread block

typedef __attribute__((ext_vector_type(16))) _Float16 v16h;
typedef __attribute__((ext_vector_type(8)))  float    v8f;

// ---- workspace layout (bytes) ----
#define OFF_FRAGS   0                                   // 8*32*16 halfs = 8192 B
#define OFF_FRAGE   8192
#define OFF_HIST1   16384                               // 128*32768 u32 = 16 MB
#define OFF_HIST2   (OFF_HIST1 + 128*32768*4)           // 128*256 u32
#define OFF_ROWSUM  (OFF_HIST2 + 128*256*4)             // 128 f32
#define OFF_CCOUNT  (OFF_ROWSUM + 512)                  // 128 u32
#define OFF_T1      (OFF_CCOUNT + 512)                  // 128 u32
#define OFF_CB1     (OFF_T1 + 512)
#define OFF_PREFIX  (OFF_CB1 + 512)
#define OFF_NEED    (OFF_PREFIX + 512)
#define OFF_CBITS   (OFF_NEED + 512)                    // 128*CAP u32
#define OFF_CIDX    (OFF_CBITS + 128*CAP*4)
#define OFF_CVAL    (OFF_CIDX + 128*CAP*4)
// zero region covers hist1 + hist2 + rowsum + ccount (contiguous)
#define ZERO_WORDS  ((OFF_T1 - OFF_HIST1) / 4)

__global__ void zero_kernel(unsigned* p, unsigned long long n) {
    unsigned long long i = (unsigned long long)blockIdx.x * blockDim.x + threadIdx.x;
    unsigned long long stride = (unsigned long long)gridDim.x * blockDim.x;
    for (; i < n; i += stride) p[i] = 0u;
}

// Kernel 1: S = x[:,1:] @ rp^T, E = iid_weight[iid]; pack both into f16 WMMA
// A-fragments (contiguous K layout, matching B): lane half h, element e -> K = 16h+e
__global__ void setup_kernel(const float* __restrict__ x,
                             const float* __restrict__ iidW,
                             const float* __restrict__ rpW,
                             _Float16* __restrict__ fragS,
                             _Float16* __restrict__ fragE) {
    __shared__ float Sm[BATCH * EMB];
    __shared__ float Em[BATCH * EMB];
    int tid = threadIdx.x;  // 256 threads
    for (int i = tid; i < BATCH * EMB; i += 256) {
        int b = i >> 5, d = i & 31;
        const float* xr = x + b * 33;
        float s = 0.f;
        #pragma unroll
        for (int j = 0; j < EMB; ++j) s += xr[1 + j] * rpW[d * EMB + j];
        Sm[i] = s;
        int ii = (int)xr[0];
        Em[i] = iidW[ii * EMB + d];
    }
    __syncthreads();
    int t = tid >> 5, l = tid & 31;
    int m = t * 16 + (l & 15);
    int half = l >> 4;
    #pragma unroll
    for (int e = 0; e < 16; ++e) {
        int k = half * 16 + e;
        fragS[(t * 32 + l) * 16 + e] = (_Float16)Sm[m * EMB + k];
        fragE[(t * 32 + l) * 16 + e] = (_Float16)Em[m * EMB + k];
    }
}

// The fused WMMA pass. mode 0: hist1 of bits>>16. mode 1: hist2 of (bits>>8)&0xFF
// within the row's level-1 bin. mode 2: sum A below prefix + collect boundary candidates.
__global__ __launch_bounds__(256) void gemm_pass(
    const float* __restrict__ U,
    const _Float16* __restrict__ fragS, const _Float16* __restrict__ fragE,
    unsigned* __restrict__ hist1, unsigned* __restrict__ hist2,
    const unsigned* __restrict__ T1v, const unsigned* __restrict__ prefv,
    float* __restrict__ rowsum, unsigned* __restrict__ ccount,
    unsigned* __restrict__ cbits, unsigned* __restrict__ cidx,
    float* __restrict__ cval, int mode) {
    __shared__ float rowAcc[BATCH];
    if (threadIdx.x < BATCH) rowAcc[threadIdx.x] = 0.f;
    __syncthreads();

    int wave = threadIdx.x >> 5, lane = threadIdx.x & 31;
    int tile = blockIdx.x * 8 + wave;
    if (tile < NTILES) {   // wave-uniform: EXEC all-ones inside (WMMA requirement)
        int n = tile * 16 + (lane & 15);
        int half = lane >> 4;
        // B fragment: 32x16 (KxN); lane half h holds K = 16h + e, contiguous in memory
        const float* up = U + (size_t)n * EMB + half * 16;
        float4 u0 = *(const float4*)(up);
        float4 u1 = *(const float4*)(up + 4);
        float4 u2 = *(const float4*)(up + 8);
        float4 u3 = *(const float4*)(up + 12);
        v16h b;
        b[0]  = (_Float16)u0.x; b[1]  = (_Float16)u0.y; b[2]  = (_Float16)u0.z; b[3]  = (_Float16)u0.w;
        b[4]  = (_Float16)u1.x; b[5]  = (_Float16)u1.y; b[6]  = (_Float16)u1.z; b[7]  = (_Float16)u1.w;
        b[8]  = (_Float16)u2.x; b[9]  = (_Float16)u2.y; b[10] = (_Float16)u2.z; b[11] = (_Float16)u2.w;
        b[12] = (_Float16)u3.x; b[13] = (_Float16)u3.y; b[14] = (_Float16)u3.z; b[15] = (_Float16)u3.w;

        for (int t = 0; t < MTILES; ++t) {
            v16h a = *(const v16h*)(fragS + (t * 32 + lane) * 16);
            v8f cS = {};
            cS = __builtin_amdgcn_wmma_f32_16x16x32_f16(
                false, a, false, b, (short)0, cS, false, false);
            v8f cE = {};
            if (mode == 2) {
                v16h ae = *(const v16h*)(fragE + (t * 32 + lane) * 16);
                cE = __builtin_amdgcn_wmma_f32_16x16x32_f16(
                    false, ae, false, b, (short)0, cE, false, false);
            }
            #pragma unroll
            for (int j = 0; j < 8; ++j) {
                int m = t * 16 + j + half * 8;       // C/D layout: VGPR j, lane-half adds 8
                float p = fabsf(cS[j] - 4.0f);
                unsigned bits = __float_as_uint(p);  // positive float: bits monotone in value
                if (mode == 0) {
                    atomicAdd(&hist1[m * 32768 + (int)(bits >> 16)], 1u);
                } else if (mode == 1) {
                    if ((bits >> 16) == T1v[m])
                        atomicAdd(&hist2[m * 256 + (int)((bits >> 8) & 0xFFu)], 1u);
                } else {
                    unsigned key = bits >> 8;
                    unsigned pr  = prefv[m];
                    float v = (key < pr) ? cE[j] : 0.f;
                    // reduce over the 16-lane half (n varies across lanes, m fixed)
                    for (int off = 1; off < 16; off <<= 1) v += __shfl_xor(v, off, 32);
                    if ((lane & 15) == 0) atomicAdd(&rowAcc[m], v);
                    if (key == pr) {
                        unsigned slot = atomicAdd(&ccount[m], 1u);
                        if (slot < CAP) {
                            cbits[m * CAP + slot] = bits;
                            cidx [m * CAP + slot] = (unsigned)n;
                            cval [m * CAP + slot] = cE[j];
                        }
                    }
                }
            }
        }
    }
    __syncthreads();
    if (mode == 2 && threadIdx.x < BATCH)
        atomicAdd(&rowsum[threadIdx.x], rowAcc[threadIdx.x]);
}

// Per-row prefix scans over the histograms to locate the 50000th smallest.
__global__ void scan_kernel(const unsigned* __restrict__ hist1,
                            const unsigned* __restrict__ hist2,
                            unsigned* T1, unsigned* cb1,
                            unsigned* pref, unsigned* need, int mode) {
    int r = threadIdx.x;
    if (r >= BATCH) return;
    if (mode == 0) {
        unsigned cum = 0;
        for (int b = 0; b < 32768; ++b) {
            unsigned c = hist1[r * 32768 + b];
            if (cum + c >= TOPK) { T1[r] = (unsigned)b; cb1[r] = cum; return; }
            cum += c;
        }
        T1[r] = 32767u; cb1[r] = cum;
    } else {
        unsigned cum = cb1[r], t1 = T1[r];
        for (int b = 0; b < 256; ++b) {
            unsigned c = hist2[r * 256 + b];
            if (cum + c >= TOPK) {
                pref[r] = (t1 << 8) | (unsigned)b;
                need[r] = TOPK - cum;
                return;
            }
            cum += c;
        }
        pref[r] = (t1 << 8) | 255u; need[r] = (TOPK > cum) ? (TOPK - cum) : 0u;
    }
}

// Pick the `need` smallest boundary candidates per row, tie-break by user index
// (matches top_k's smallest-index-first), finish the mean.
__global__ void final_kernel(const float* __restrict__ rowsum,
                             const unsigned* __restrict__ need,
                             const unsigned* __restrict__ ccount,
                             unsigned* __restrict__ cbits,
                             const unsigned* __restrict__ cidx,
                             const float* __restrict__ cval,
                             float* __restrict__ out) {
    int r = threadIdx.x;
    if (r >= BATCH) return;
    unsigned cnt = ccount[r]; if (cnt > CAP) cnt = CAP;
    unsigned nd = need[r];    if (nd > cnt) nd = cnt;
    unsigned base = (unsigned)r * CAP;
    float s = rowsum[r];
    for (unsigned sel = 0; sel < nd; ++sel) {
        unsigned bb = 0xFFFFFFFFu, bi = 0xFFFFFFFFu; int best = -1;
        for (unsigned i = 0; i < cnt; ++i) {
            unsigned v = cbits[base + i];
            if (v == 0xFFFFFFFFu) continue;  // consumed marker
            unsigned id = cidx[base + i];
            if (v < bb || (v == bb && id < bi)) { bb = v; bi = id; best = (int)i; }
        }
        if (best < 0) break;
        s += cval[base + (unsigned)best];
        cbits[base + (unsigned)best] = 0xFFFFFFFFu;
    }
    out[r] = s / (float)TOPK;
}

extern "C" void kernel_launch(void* const* d_in, const int* in_sizes, int n_in,
                              void* d_out, int out_size, void* d_ws, size_t ws_size,
                              hipStream_t stream) {
    (void)in_sizes; (void)n_in; (void)out_size; (void)ws_size;
    const float* x    = (const float*)d_in[0];   // [128, 33]
    const float* uidW = (const float*)d_in[1];   // [200000, 32]
    const float* iidW = (const float*)d_in[2];   // [50000, 32]
    const float* rpW  = (const float*)d_in[3];   // [32, 32]
    float* out = (float*)d_out;                  // [128] f32

    char* ws = (char*)d_ws;
    _Float16* fragS  = (_Float16*)(ws + OFF_FRAGS);
    _Float16* fragE  = (_Float16*)(ws + OFF_FRAGE);
    unsigned* hist1  = (unsigned*)(ws + OFF_HIST1);
    unsigned* hist2  = (unsigned*)(ws + OFF_HIST2);
    float*    rowsum = (float*)   (ws + OFF_ROWSUM);
    unsigned* ccount = (unsigned*)(ws + OFF_CCOUNT);
    unsigned* T1     = (unsigned*)(ws + OFF_T1);
    unsigned* cb1    = (unsigned*)(ws + OFF_CB1);
    unsigned* pref   = (unsigned*)(ws + OFF_PREFIX);
    unsigned* needv  = (unsigned*)(ws + OFF_NEED);
    unsigned* cbits  = (unsigned*)(ws + OFF_CBITS);
    unsigned* cidx   = (unsigned*)(ws + OFF_CIDX);
    float*    cval   = (float*)   (ws + OFF_CVAL);

    // 0) zero counters (hist1+hist2+rowsum+ccount, contiguous)
    zero_kernel<<<2048, 256, 0, stream>>>(hist1, (unsigned long long)ZERO_WORDS);
    // 1) build f16 A-fragments for S and E
    setup_kernel<<<1, 256, 0, stream>>>(x, iidW, rpW, fragS, fragE);
    // 2) WMMA pass: level-1 histogram (bits>>16)
    gemm_pass<<<GEMM_BLOCKS, 256, 0, stream>>>(uidW, fragS, fragE, hist1, hist2,
                                               T1, pref, rowsum, ccount,
                                               cbits, cidx, cval, 0);
    // 3) scan level-1
    scan_kernel<<<1, 128, 0, stream>>>(hist1, hist2, T1, cb1, pref, needv, 0);
    // 4) WMMA pass: level-2 histogram within the k-th bin
    gemm_pass<<<GEMM_BLOCKS, 256, 0, stream>>>(uidW, fragS, fragE, hist1, hist2,
                                               T1, pref, rowsum, ccount,
                                               cbits, cidx, cval, 1);
    // 5) scan level-2 -> 24-bit prefix threshold + residual count
    scan_kernel<<<1, 128, 0, stream>>>(hist1, hist2, T1, cb1, pref, needv, 1);
    // 6) WMMA pass: sum A below threshold, collect boundary candidates
    gemm_pass<<<GEMM_BLOCKS, 256, 0, stream>>>(uidW, fragS, fragE, hist1, hist2,
                                               T1, pref, rowsum, ccount,
                                               cbits, cidx, cval, 2);
    // 7) resolve boundary + mean
    final_kernel<<<1, 128, 0, stream>>>(rowsum, needv, ccount, cbits, cidx, cval, out);
}
